// AdjacencyLayer_17978733101166
// MI455X (gfx1250) — compile-verified, matrix operand order checked by reading, not run on hardware
//
#include <hip/hip_runtime.h>
#include <math.h>

// ---- problem constants (from reference: b=8192, d=256) ----
#define B_ROWS 8192
#define DDIM   256
#define KTOT   512   // [x | e] concatenated along K

typedef __attribute__((ext_vector_type(16))) __bf16 v16bf;
typedef __attribute__((ext_vector_type(8)))  __bf16 v8bf;
typedef __attribute__((ext_vector_type(8)))  float  v8f;

__device__ __forceinline__ unsigned short f32_to_bf16_rn(float f) {
    unsigned int u = __float_as_uint(f);
    unsigned int r = u + 0x7FFFu + ((u >> 16) & 1u);  // round-to-nearest-even
    return (unsigned short)(r >> 16);
}
__device__ __forceinline__ float bf16u_to_f32(unsigned short h) {
    return __uint_as_float(((unsigned int)h) << 16);
}

// Kernel 1: build split-bf16 operands.
//   A[i, 0:256]   = c1 * x[i,:]   (c1 = -2(1-p)/d)
//   A[i, 256:512] = c2 * e[i,:]   (c2 = p/d, e = exp(1-dc))
//   B[i, 0:256]   = x[i,:],  B[i, 256:512] = e[i,:]
// each stored as bf16 hi + bf16 lo (lo = bf16(v - f32(hi))).
__global__ void prep_split_kernel(const float* __restrict__ x,
                                  const float* __restrict__ dc,
                                  const float* __restrict__ pptr,
                                  unsigned short* __restrict__ Ahi,
                                  unsigned short* __restrict__ Alo,
                                  unsigned short* __restrict__ Bhi,
                                  unsigned short* __restrict__ Blo) {
    int idx = blockIdx.x * blockDim.x + threadIdx.x;   // 0 .. 8192*256-1
    int i = idx >> 8;
    int k = idx & 255;
    float p  = pptr[0];
    float c1 = -2.0f * (1.0f - p) / (float)DDIM;
    float c2 = p / (float)DDIM;

    float xv = x[idx];
    float ev = expf(1.0f - dc[idx]);

    size_t bx = (size_t)i * KTOT + k;          // x half
    size_t be = bx + DDIM;                     // e half

    // B (unscaled)
    unsigned short h = f32_to_bf16_rn(xv);
    Bhi[bx] = h; Blo[bx] = f32_to_bf16_rn(xv - bf16u_to_f32(h));
    h = f32_to_bf16_rn(ev);
    Bhi[be] = h; Blo[be] = f32_to_bf16_rn(ev - bf16u_to_f32(h));

    // A (scaled)
    float av = c1 * xv;
    h = f32_to_bf16_rn(av);
    Ahi[bx] = h; Alo[bx] = f32_to_bf16_rn(av - bf16u_to_f32(h));
    av = c2 * ev;
    h = f32_to_bf16_rn(av);
    Ahi[be] = h; Alo[be] = f32_to_bf16_rn(av - bf16u_to_f32(h));
}

// Kernel 2: sq[i] = sum_k x[i,k]^2  (one 256-thread block per row)
__global__ void rowsq_kernel(const float* __restrict__ x, float* __restrict__ sq) {
    __shared__ float red[256];
    int i = blockIdx.x;
    float v = x[(size_t)i * DDIM + threadIdx.x];
    red[threadIdx.x] = v * v;
    __syncthreads();
    for (int s = 128; s > 0; s >>= 1) {
        if (threadIdx.x < (unsigned)s) red[threadIdx.x] += red[threadIdx.x + s];
        __syncthreads();
    }
    if (threadIdx.x == 0) sq[i] = red[0];
}

// Load a 16x32 bf16 WMMA A/B fragment from a row-major [rows][KTOT] bf16 array.
// Lane l<16 holds row (base + l), K = {kk..kk+7, kk+16..kk+23};
// lane l>=16 holds row (base + l-16), K = {kk+8..kk+15, kk+24..kk+31}.
// Both runs are contiguous 16-byte loads.
__device__ __forceinline__ v16bf load_frag(const unsigned short* __restrict__ mat,
                                           int rowBase, int kk, int lr, int khalf) {
    const unsigned short* base = mat + (size_t)(rowBase + lr) * KTOT + kk + khalf;
    v8bf lo16 = *(const v8bf*)(base);
    v8bf hi16 = *(const v8bf*)(base + 16);
    return __builtin_shufflevector(lo16, hi16,
                                   0, 1, 2, 3, 4, 5, 6, 7,
                                   8, 9, 10, 11, 12, 13, 14, 15);
}

// Kernel 3: fused GEMM + epilogue.
// Block = 256 threads = 8 wave32s; block tile 128x128; wave tile 32x64.
// acc[i,j] = c1*x_i.x_j + c2*e_i.e_j via split-bf16 (hi*hi + hi*lo + lo*hi).
// out = 1 / (acc + q*(sq_i+sq_j)), diagonal forced to 1.
__global__ __launch_bounds__(256)
void adjacency_wmma_kernel(const unsigned short* __restrict__ Ahi,
                           const unsigned short* __restrict__ Alo,
                           const unsigned short* __restrict__ Bhi,
                           const unsigned short* __restrict__ Blo,
                           const float* __restrict__ sq,
                           const float* __restrict__ pptr,
                           float* __restrict__ out) {
    const int lane  = threadIdx.x & 31;
    const int wave  = threadIdx.x >> 5;
    const int wm    = wave >> 1;                 // 0..3  (M direction)
    const int wn    = wave & 1;                  // 0..1  (N direction)
    const int rowBase = blockIdx.y * 128 + wm * 32;   // wave M start
    const int colBase = blockIdx.x * 128 + wn * 64;   // wave N start
    const int lr    = lane & 15;
    const int khalf = (lane >> 4) << 3;          // 0 or 8

    v8f acc[2][4];
#pragma unroll
    for (int mt = 0; mt < 2; ++mt)
#pragma unroll
        for (int nt = 0; nt < 4; ++nt)
            acc[mt][nt] = (v8f)(0.0f);

    for (int kk = 0; kk < KTOT; kk += 32) {
        v16bf ahi[2], alo[2], bhi[4], blo[4];
#pragma unroll
        for (int mt = 0; mt < 2; ++mt) {
            ahi[mt] = load_frag(Ahi, rowBase + mt * 16, kk, lr, khalf);
            alo[mt] = load_frag(Alo, rowBase + mt * 16, kk, lr, khalf);
        }
#pragma unroll
        for (int nt = 0; nt < 4; ++nt) {
            bhi[nt] = load_frag(Bhi, colBase + nt * 16, kk, lr, khalf);
            blo[nt] = load_frag(Blo, colBase + nt * 16, kk, lr, khalf);
        }
#pragma unroll
        for (int mt = 0; mt < 2; ++mt) {
#pragma unroll
            for (int nt = 0; nt < 4; ++nt) {
                acc[mt][nt] = __builtin_amdgcn_wmma_f32_16x16x32_bf16(
                    false, ahi[mt], false, bhi[nt], (short)0, acc[mt][nt], false, false);
                acc[mt][nt] = __builtin_amdgcn_wmma_f32_16x16x32_bf16(
                    false, ahi[mt], false, blo[nt], (short)0, acc[mt][nt], false, false);
                acc[mt][nt] = __builtin_amdgcn_wmma_f32_16x16x32_bf16(
                    false, alo[mt], false, bhi[nt], (short)0, acc[mt][nt], false, false);
            }
        }
    }

    const float p = pptr[0];
    const float q = (1.0f - p) / (float)DDIM;
    const int rOff = khalf;                      // C/D layout: lanes>=16 are M=r+8

#pragma unroll
    for (int mt = 0; mt < 2; ++mt) {
#pragma unroll
        for (int nt = 0; nt < 4; ++nt) {
            const int col  = colBase + nt * 16 + lr;
            const float sc = sq[col];
#pragma unroll
            for (int r = 0; r < 8; ++r) {
                const int row = rowBase + mt * 16 + r + rOff;
                const float denom = acc[mt][nt][r] + q * (sq[row] + sc);
                out[(size_t)row * B_ROWS + col] = (row == col) ? 1.0f : (1.0f / denom);
            }
        }
    }
}

extern "C" void kernel_launch(void* const* d_in, const int* in_sizes, int n_in,
                              void* d_out, int out_size, void* d_ws, size_t ws_size,
                              hipStream_t stream) {
    const float* x    = (const float*)d_in[0];
    const float* dc   = (const float*)d_in[1];
    const float* parm = (const float*)d_in[2];
    float* out        = (float*)d_out;

    // workspace layout: Ahi | Alo | Bhi | Blo (bf16, 8192x512 each) | sq (f32, 8192)
    const size_t matElems = (size_t)B_ROWS * KTOT;
    unsigned short* Ahi = (unsigned short*)d_ws;
    unsigned short* Alo = Ahi + matElems;
    unsigned short* Bhi = Alo + matElems;
    unsigned short* Blo = Bhi + matElems;
    float* sqv          = (float*)(Blo + matElems);

    const int prepBlocks = (B_ROWS * DDIM) / 256;
    prep_split_kernel<<<prepBlocks, 256, 0, stream>>>(x, dc, parm, Ahi, Alo, Bhi, Blo);
    rowsq_kernel<<<B_ROWS, 256, 0, stream>>>(x, sqv);

    dim3 grid(B_ROWS / 128, B_ROWS / 128);
    adjacency_wmma_kernel<<<grid, 256, 0, stream>>>(Ahi, Alo, Bhi, Blo, sqv, parm, out);
}
